// ConvTransformerEncoderLayer_5411658793436
// MI455X (gfx1250) — matrix operand order repaired and checked by reading, not compiled, not run on hardware
//
#include <hip/hip_runtime.h>
#include <hip/hip_bf16.h>
#include <math.h>

typedef __attribute__((ext_vector_type(16))) __bf16 v16bf;
typedef __attribute__((ext_vector_type(8)))  float  v8f;

#define LSEQ   256
#define DMODEL 256
#define NHEADS 8
#define PDIM   64
#define HTRI   4
#define DKTRI  16
#define LL     (LSEQ*LSEQ)
#define NEGINF (-1e9f)

// ---- WMMA fragment index helpers (contiguous k = 16*(lane/16) + e, same order for A and B) ----
// A (16x32 bf16): lanes 0-15 -> K0..15, lanes 16-31 -> K16..31
__device__ __forceinline__ int fka(int lane, int e) {
  return ((lane >> 4) << 4) + e;
}
// B (32x16 bf16): lanes 0-15 -> K0..15, lanes 16-31 -> K16..31
__device__ __forceinline__ int fkb(int lane, int e) {
  return ((lane >> 4) << 4) + e;
}
__device__ __forceinline__ __bf16 tobf(float f) { return (__bf16)f; }
__device__ __forceinline__ float sigmoidf_(float x) { return 1.0f / (1.0f + __expf(-x)); }

// =====================================================================
// LayerNorm over last dim, one wave per row
// =====================================================================
__global__ void layernorm_kernel(const float* __restrict__ x, const float* __restrict__ g,
                                 const float* __restrict__ b, float* __restrict__ y, int dim) {
  int row = blockIdx.x;
  const float* xr = x + (size_t)row * dim;
  float s = 0.f, ss = 0.f;
  for (int c = threadIdx.x; c < dim; c += 32) { float v = xr[c]; s += v; ss += v * v; }
  for (int o = 16; o > 0; o >>= 1) { s += __shfl_xor(s, o, 32); ss += __shfl_xor(ss, o, 32); }
  float mean = s / dim;
  float var  = ss / dim - mean * mean;
  float inv  = rsqrtf(var + 1e-5f);
  float* yr = y + (size_t)row * dim;
  for (int c = threadIdx.x; c < dim; c += 32)
    yr[c] = (xr[c] - mean) * inv * g[c] + b[c];
}

// =====================================================================
// Generic GEMM: C = act(A[MxK] @ W[KxN] + bias) (+ res)
// 256 threads (8 waves), 64x64 tile, double-buffered LDS bf16 staging,
// WMMA 16x16x32.  act: 0=none, 1=relu, 2=sigmoid
// BOUNDED=false: M%64==0, N%64==0, K%32==0 (no guards, vector A loads)
// =====================================================================
template <bool BOUNDED>
__global__ void gemm_kernel(const float* __restrict__ A, int lda,
                            const float* __restrict__ W, int ldw,
                            const float* __restrict__ bias,
                            const float* __restrict__ res,
                            float* __restrict__ C, int ldc,
                            int M, int N, int K, int act) {
  __shared__ __bf16 As[2][64][40];   // [buf][m][k]
  __shared__ __bf16 Bs[2][64][40];   // [buf][n][k] (W transposed while staging)
  int tid = threadIdx.x;
  int lane = tid & 31, wave = tid >> 5;
  int m0 = blockIdx.x * 64, n0 = blockIdx.y * 64;
  int mloc = (wave & 3) * 16, nloc = (wave >> 2) * 32;
  int sr = tid >> 2, skq = (tid & 3) * 8;

  auto stage = [&](int buf, int k0) {
    if (!BOUNDED) {
      // aligned fast path: vector A loads, straight-line B loads
      const float4* ap =
          reinterpret_cast<const float4*>(A + (size_t)(m0 + sr) * lda + k0 + skq);
      float4 a0 = ap[0], a1 = ap[1];
      As[buf][sr][skq + 0] = tobf(a0.x); As[buf][sr][skq + 1] = tobf(a0.y);
      As[buf][sr][skq + 2] = tobf(a0.z); As[buf][sr][skq + 3] = tobf(a0.w);
      As[buf][sr][skq + 4] = tobf(a1.x); As[buf][sr][skq + 5] = tobf(a1.y);
      As[buf][sr][skq + 6] = tobf(a1.z); As[buf][sr][skq + 7] = tobf(a1.w);
#pragma unroll
      for (int e = 0; e < 8; e++)
        Bs[buf][sr][skq + e] = tobf(W[(size_t)(k0 + skq + e) * ldw + n0 + sr]);
      // speculative prefetch of the tile after next (gfx1250 global_prefetch)
      __builtin_prefetch(A + (size_t)(m0 + sr) * lda + k0 + skq + 64, 0, 0);
    } else {
      // branch-free guarded path: clamp address, select zero
      int rA = (m0 + sr < M) ? (m0 + sr) : (M - 1);
      int nB = (n0 + sr < N) ? (n0 + sr) : (N - 1);
#pragma unroll
      for (int e = 0; e < 8; e++) {
        int k  = k0 + skq + e;
        int kc = (k < K) ? k : (K - 1);
        float va = A[(size_t)rA * lda + kc];
        float vb = W[(size_t)kc * ldw + nB];
        bool okk = (k < K);
        As[buf][sr][skq + e] = tobf((okk && (m0 + sr < M)) ? va : 0.f);
        Bs[buf][sr][skq + e] = tobf((okk && (n0 + sr < N)) ? vb : 0.f);
      }
    }
  };

  v8f acc0 = {}; v8f acc1 = {};
  int nk = (K + 31) / 32;
  stage(0, 0);
  for (int kt = 0; kt < nk; kt++) {
    __syncthreads();
    if (kt + 1 < nk) stage((kt + 1) & 1, (kt + 1) * 32);
    int buf = kt & 1;
    v16bf a, b0, b1;
    int mm = mloc + (lane & 15);
    int nn = lane & 15;
#pragma unroll
    for (int e = 0; e < 16; e++) {
      a[e]  = As[buf][mm][fka(lane, e)];
      b0[e] = Bs[buf][nloc + nn][fkb(lane, e)];
      b1[e] = Bs[buf][nloc + 16 + nn][fkb(lane, e)];
    }
    acc0 = __builtin_amdgcn_wmma_f32_16x16x32_bf16(false, a, false, b0, (short)0, acc0, false, false);
    acc1 = __builtin_amdgcn_wmma_f32_16x16x32_bf16(false, a, false, b1, (short)0, acc1, false, false);
  }
  int hi = lane >> 4;
#pragma unroll
  for (int t = 0; t < 2; t++) {
    v8f av = t ? acc1 : acc0;
    int col = n0 + nloc + t * 16 + (lane & 15);
#pragma unroll
    for (int rr = 0; rr < 8; rr++) {
      int row = m0 + mloc + rr + hi * 8;
      if (!BOUNDED || (row < M && col < N)) {
        float v = av[rr];
        if (bias) v += bias[col];
        if (act == 1) v = fmaxf(v, 0.f);
        else if (act == 2) v = sigmoidf_(v);
        if (res) v += res[(size_t)row * ldc + col];
        C[(size_t)row * ldc + col] = v;
      }
    }
  }
}

// =====================================================================
// out = a * sigmoid(g) [* mask_i * mask_j]   over (L,L,P) flat
// =====================================================================
__global__ void gate_kernel(float* __restrict__ out, const float* __restrict__ a,
                            const float* __restrict__ g, const float* __restrict__ mask,
                            long total) {
  long idx = (long)blockIdx.x * blockDim.x + threadIdx.x;
  if (idx >= total) return;
  float v = a[idx] * sigmoidf_(g[idx]);
  if (mask) {
    long ij = idx / PDIM;
    int j = (int)(ij % LSEQ), i = (int)(ij / LSEQ);
    v *= mask[i] * mask[j];
  }
  out[idx] = v;
}

// =====================================================================
// Triangle multiplicative einsum, one channel per block.z, 2x2 tiles:
// OUTGOING: O[i,j,d] = sum_k Lt[i,k,d]*Rt[j,k,d]
// ingoing : O[i,j,d] = sum_k Lt[k,i,d]*Rt[k,j,d]
// One wave computes a 32x32 output tile (4 wmma per k-step, 1 frag/wmma).
// =====================================================================
template <bool OUTGOING>
__global__ void tri_einsum_kernel(const float* __restrict__ Lt, const float* __restrict__ Rt,
                                  float* __restrict__ Ot) {
  int it = blockIdx.x, jt = blockIdx.y, d = blockIdx.z;
  int lane = threadIdx.x, l15 = lane & 15, hi = lane >> 4;
  int m0 = it * 32, n0 = jt * 32;
  v8f acc00 = {}, acc01 = {}, acc10 = {}, acc11 = {};
  for (int k0 = 0; k0 < LSEQ; k0 += 32) {
    v16bf a0, a1, b0, b1;
#pragma unroll
    for (int e = 0; e < 16; e++) {
      int ka = k0 + fka(lane, e);
      int kb = k0 + fkb(lane, e);
      size_t ia0 = OUTGOING ? ((size_t)(m0 + l15) * LSEQ + ka)      : ((size_t)ka * LSEQ + m0 + l15);
      size_t ia1 = OUTGOING ? ((size_t)(m0 + 16 + l15) * LSEQ + ka) : ((size_t)ka * LSEQ + m0 + 16 + l15);
      size_t ib0 = OUTGOING ? ((size_t)(n0 + l15) * LSEQ + kb)      : ((size_t)kb * LSEQ + n0 + l15);
      size_t ib1 = OUTGOING ? ((size_t)(n0 + 16 + l15) * LSEQ + kb) : ((size_t)kb * LSEQ + n0 + 16 + l15);
      a0[e] = tobf(Lt[ia0 * PDIM + d]);
      a1[e] = tobf(Lt[ia1 * PDIM + d]);
      b0[e] = tobf(Rt[ib0 * PDIM + d]);
      b1[e] = tobf(Rt[ib1 * PDIM + d]);
    }
    acc00 = __builtin_amdgcn_wmma_f32_16x16x32_bf16(false, a0, false, b0, (short)0, acc00, false, false);
    acc01 = __builtin_amdgcn_wmma_f32_16x16x32_bf16(false, a0, false, b1, (short)0, acc01, false, false);
    acc10 = __builtin_amdgcn_wmma_f32_16x16x32_bf16(false, a1, false, b0, (short)0, acc10, false, false);
    acc11 = __builtin_amdgcn_wmma_f32_16x16x32_bf16(false, a1, false, b1, (short)0, acc11, false, false);
  }
  auto store = [&](const v8f& acc, int mb, int nb) {
    int col = nb + l15;
#pragma unroll
    for (int rr = 0; rr < 8; rr++) {
      int row = mb + rr + hi * 8;
      Ot[((size_t)row * LSEQ + col) * PDIM + d] = acc[rr];
    }
  };
  store(acc00, m0, n0);
  store(acc01, m0, n0 + 16);
  store(acc10, m0 + 16, n0);
  store(acc11, m0 + 16, n0 + 16);
}

// =====================================================================
// OPM helpers
// =====================================================================
__global__ void wd2_permute_kernel(const float* __restrict__ Wd2, float* __restrict__ W2r) {
  int idx = blockIdx.x * blockDim.x + threadIdx.x;  // d*2048 + c*64 + p
  if (idx >= 32 * 32 * 64) return;
  int p = idx & 63, c = (idx >> 6) & 31, d = idx >> 11;
  W2r[idx] = Wd2[(size_t)(c * 32 + d) * 64 + p];
}

// pair[i,j,p] = pairwise[i,j,p] + sum_c s[j,c]*T[i,c*64+p] + bd2[p]
__global__ void opm_finish_kernel(const float* __restrict__ s, const float* __restrict__ T,
                                  const float* __restrict__ pw, const float* __restrict__ bd2,
                                  float* __restrict__ pair) {
  int i = blockIdx.x, jt = blockIdx.y;
  int lane = threadIdx.x, hi = lane >> 4, l15 = lane & 15;
  v16bf a;
  int mm = jt * 16 + l15;
#pragma unroll
  for (int e = 0; e < 16; e++) a[e] = tobf(s[(size_t)mm * 32 + fka(lane, e)]);
#pragma unroll
  for (int nt = 0; nt < 4; nt++) {
    v16bf b;
#pragma unroll
    for (int e = 0; e < 16; e++)
      b[e] = tobf(T[(size_t)i * 2048 + (size_t)fkb(lane, e) * 64 + nt * 16 + l15]);
    v8f acc = {};
    acc = __builtin_amdgcn_wmma_f32_16x16x32_bf16(false, a, false, b, (short)0, acc, false, false);
    int col = nt * 16 + l15;
#pragma unroll
    for (int rr = 0; rr < 8; rr++) {
      int j = jt * 16 + rr + hi * 8;
      size_t o = ((size_t)i * LSEQ + j) * PDIM + col;
      pair[o] = pw[o] + acc[rr] + bd2[col];
    }
  }
}

// =====================================================================
// Wave-level masked softmax over 16 rows x 256 cols in LDS.
// 2 lanes per row, 128 cols each, combine via shfl_xor.
// =====================================================================
__device__ __forceinline__ void softmax16x256(float (*sm)[256], int lane) {
  int r = lane >> 1, half = (lane & 1) * 128;
  float mx = -3.4e38f;
  for (int j = 0; j < 128; j++) mx = fmaxf(mx, sm[r][half + j]);
  mx = fmaxf(mx, __shfl_xor(mx, 1, 32));
  float ssum = 0.f;
  for (int j = 0; j < 128; j++) {
    float e = __expf(sm[r][half + j] - mx);
    sm[r][half + j] = e;
    ssum += e;
  }
  ssum += __shfl_xor(ssum, 1, 32);
  float inv = 1.f / ssum;
  for (int j = 0; j < 128; j++) sm[r][half + j] *= inv;
}

// =====================================================================
// Sequence MHA: one wave per (head, 16-query tile). DK=32 (one k-step).
// =====================================================================
__global__ void mha_kernel(const float* __restrict__ Q, const float* __restrict__ Km,
                           const float* __restrict__ V, const float* __restrict__ bproj,
                           const float* __restrict__ mask, float* __restrict__ O) {
  __shared__ float sm[16][256];
  int h = blockIdx.x, it = blockIdx.y;
  int lane = threadIdx.x, hi = lane >> 4, l15 = lane & 15;
  int i0 = it * 16;
  const float invs = 0.17677669529663687f;  // 1/sqrt(32)
  v16bf a;
#pragma unroll
  for (int e = 0; e < 16; e++)
    a[e] = tobf(Q[(size_t)(i0 + l15) * DMODEL + h * 32 + fka(lane, e)]);
  for (int jt = 0; jt < 16; jt++) {
    v16bf b;
#pragma unroll
    for (int e = 0; e < 16; e++)
      b[e] = tobf(Km[(size_t)(jt * 16 + l15) * DMODEL + h * 32 + fkb(lane, e)]);
    v8f acc = {};
    acc = __builtin_amdgcn_wmma_f32_16x16x32_bf16(false, a, false, b, (short)0, acc, false, false);
    int j = jt * 16 + l15;
    float mj = mask[j];
#pragma unroll
    for (int rr = 0; rr < 8; rr++) {
      int row = rr + hi * 8;
      float v = acc[rr] * invs + bproj[((size_t)(i0 + row) * LSEQ + j) * NHEADS + h];
      sm[row][j] = (mj > 0.f) ? v : NEGINF;
    }
  }
  __syncthreads();
  softmax16x256(sm, lane);
  __syncthreads();
#pragma unroll
  for (int nt = 0; nt < 2; nt++) {
    v8f acc = {};
    for (int k0 = 0; k0 < 256; k0 += 32) {
      v16bf aa, bb;
#pragma unroll
      for (int e = 0; e < 16; e++) {
        aa[e] = tobf(sm[l15][k0 + fka(lane, e)]);
        bb[e] = tobf(V[(size_t)(k0 + fkb(lane, e)) * DMODEL + h * 32 + nt * 16 + l15]);
      }
      acc = __builtin_amdgcn_wmma_f32_16x16x32_bf16(false, aa, false, bb, (short)0, acc, false, false);
    }
#pragma unroll
    for (int rr = 0; rr < 8; rr++)
      O[(size_t)(i0 + rr + hi * 8) * DMODEL + h * 32 + nt * 16 + l15] = acc[rr];
  }
}

// =====================================================================
// Triangle attention (ROW=1: rows; 0: columns). DKT=16 zero-padded
// to K=32 branch-free (clamped address + select). 1 wave/block.
// =====================================================================
template <bool ROW>
__global__ void tri_attn_kernel(const float* __restrict__ Qb, const float* __restrict__ Kb,
                                const float* __restrict__ Vb, const float* __restrict__ bproj,
                                const float* __restrict__ mask, float* __restrict__ Ob) {
  __shared__ float sm[16][256];
  int fixed = blockIdx.x, h = blockIdx.y, it = blockIdx.z;
  int lane = threadIdx.x, hi = lane >> 4, l15 = lane & 15;
  int i0 = it * 16;
  const float invs = 0.25f;  // 1/sqrt(16)
  v16bf a;
#pragma unroll
  for (int e = 0; e < 16; e++) {
    int d = fka(lane, e);
    int i = i0 + l15;
    size_t base = ROW ? ((size_t)fixed * LSEQ + i) : ((size_t)i * LSEQ + fixed);
    float v = Qb[base * PDIM + h * DKTRI + (d & 15)];
    a[e] = tobf((d < DKTRI) ? v : 0.f);
  }
  for (int jt = 0; jt < 16; jt++) {
    v16bf b;
#pragma unroll
    for (int e = 0; e < 16; e++) {
      int d = fkb(lane, e);
      int j = jt * 16 + l15;
      size_t base = ROW ? ((size_t)fixed * LSEQ + j) : ((size_t)j * LSEQ + fixed);
      float v = Kb[base * PDIM + h * DKTRI + (d & 15)];
      b[e] = tobf((d < DKTRI) ? v : 0.f);
    }
    v8f acc = {};
    acc = __builtin_amdgcn_wmma_f32_16x16x32_bf16(false, a, false, b, (short)0, acc, false, false);
    int j = jt * 16 + l15;
    float mj = mask[j];
#pragma unroll
    for (int rr = 0; rr < 8; rr++) {
      int row = rr + hi * 8;
      float v = acc[rr] * invs + bproj[((size_t)(i0 + row) * LSEQ + j) * HTRI + h];
      sm[row][j] = (mj > 0.f) ? v : NEGINF;
    }
  }
  __syncthreads();
  softmax16x256(sm, lane);
  __syncthreads();
  v8f acc = {};
  for (int k0 = 0; k0 < 256; k0 += 32) {
    v16bf aa, bb;
#pragma unroll
    for (int e = 0; e < 16; e++) {
      aa[e] = tobf(sm[l15][k0 + fka(lane, e)]);
      int j = k0 + fkb(lane, e);
      size_t base = ROW ? ((size_t)fixed * LSEQ + j) : ((size_t)j * LSEQ + fixed);
      bb[e] = tobf(Vb[base * PDIM + h * DKTRI + l15]);
    }
    acc = __builtin_amdgcn_wmma_f32_16x16x32_bf16(false, aa, false, bb, (short)0, acc, false, false);
  }
#pragma unroll
  for (int rr = 0; rr < 8; rr++) {
    int i = i0 + rr + hi * 8;
    size_t base = ROW ? ((size_t)fixed * LSEQ + i) : ((size_t)i * LSEQ + fixed);
    Ob[base * PDIM + h * DKTRI + l15] = acc[rr];
  }
}

// =====================================================================
// Fused pair transition: out = pairIn + relu(z@W1+b1)@W2 + b2
// hidden (16x256) kept in LDS. One wave per 16 rows.
// =====================================================================
__global__ void pair_transition_kernel(const float* __restrict__ z, const float* __restrict__ W1,
                                       const float* __restrict__ b1, const float* __restrict__ W2,
                                       const float* __restrict__ b2, const float* __restrict__ pairIn,
                                       float* __restrict__ pairOut) {
  __shared__ float hbuf[16][260];
  int row0 = blockIdx.x * 16;
  int lane = threadIdx.x, hi = lane >> 4, l15 = lane & 15;
  v16bf a0, a1;
#pragma unroll
  for (int e = 0; e < 16; e++) {
    a0[e] = tobf(z[(size_t)(row0 + l15) * 64 + fka(lane, e)]);
    a1[e] = tobf(z[(size_t)(row0 + l15) * 64 + 32 + fka(lane, e)]);
  }
  for (int nt = 0; nt < 16; nt++) {
    v8f acc = {};
    v16bf b;
#pragma unroll
    for (int e = 0; e < 16; e++)
      b[e] = tobf(W1[(size_t)fkb(lane, e) * 256 + nt * 16 + l15]);
    acc = __builtin_amdgcn_wmma_f32_16x16x32_bf16(false, a0, false, b, (short)0, acc, false, false);
#pragma unroll
    for (int e = 0; e < 16; e++)
      b[e] = tobf(W1[(size_t)(32 + fkb(lane, e)) * 256 + nt * 16 + l15]);
    acc = __builtin_amdgcn_wmma_f32_16x16x32_bf16(false, a1, false, b, (short)0, acc, false, false);
    int col = nt * 16 + l15;
#pragma unroll
    for (int rr = 0; rr < 8; rr++)
      hbuf[rr + hi * 8][col] = fmaxf(acc[rr] + b1[col], 0.f);
  }
  __syncthreads();
  for (int nt = 0; nt < 4; nt++) {
    v8f acc = {};
    for (int k0 = 0; k0 < 256; k0 += 32) {
      v16bf aa, bb;
#pragma unroll
      for (int e = 0; e < 16; e++) {
        aa[e] = tobf(hbuf[l15][k0 + fka(lane, e)]);
        bb[e] = tobf(W2[(size_t)(k0 + fkb(lane, e)) * 64 + nt * 16 + l15]);
      }
      acc = __builtin_amdgcn_wmma_f32_16x16x32_bf16(false, aa, false, bb, (short)0, acc, false, false);
    }
    int col = nt * 16 + l15;
#pragma unroll
    for (int rr = 0; rr < 8; rr++) {
      size_t o = (size_t)(row0 + rr + hi * 8) * 64 + col;
      pairOut[o] = pairIn[o] + acc[rr] + b2[col];
    }
  }
}

// =====================================================================
// Host orchestration
// =====================================================================
static void gemm(hipStream_t s, const float* A, int lda, const float* W, int ldw,
                 const float* bias, const float* res, float* C, int ldc,
                 int M, int N, int K, int act) {
  dim3 g((M + 63) / 64, (N + 63) / 64);
  bool aligned = (M % 64 == 0) && (N % 64 == 0) && (K % 32 == 0);
  if (aligned)
    gemm_kernel<false><<<g, 256, 0, s>>>(A, lda, W, ldw, bias, res, C, ldc, M, N, K, act);
  else
    gemm_kernel<true><<<g, 256, 0, s>>>(A, lda, W, ldw, bias, res, C, ldc, M, N, K, act);
}
static void lnorm(hipStream_t s, const float* x, const float* g, const float* b, float* y,
                  int rows, int dim) {
  layernorm_kernel<<<rows, 32, 0, s>>>(x, g, b, y, dim);
}
static void gate(hipStream_t s, float* out, const float* a, const float* g, const float* mask) {
  long total = (long)LL * PDIM;
  gate_kernel<<<(unsigned)((total + 255) / 256), 256, 0, s>>>(out, a, g, mask, total);
}

extern "C" void kernel_launch(void* const* d_in, const int* in_sizes, int n_in,
                              void* d_out, int out_size, void* d_ws, size_t ws_size,
                              hipStream_t stream) {
  (void)in_sizes; (void)n_in; (void)out_size; (void)ws_size;
  auto IN = [&](int k) { return (const float*)d_in[k]; };
  const float* src      = IN(0);
  const float* pairwise = IN(1);
  const float* mask     = IN(2);
  const float* pw_g = IN(3); const float* pw_b = IN(4);
  const float* W_p2h = IN(5);
  const float* Wq = IN(6); const float* Wk = IN(7); const float* Wv = IN(8); const float* Wo = IN(9);
  const float* bq = IN(10); const float* bk = IN(11); const float* bv = IN(12); const float* bo = IN(13);
  const float* n1g = IN(14); const float* n1b = IN(15); const float* n2g = IN(16); const float* n2b = IN(17);
  const float* stW1 = IN(18); const float* stb1 = IN(19); const float* stW2 = IN(20); const float* stb2 = IN(21);
  const float* Wd1 = IN(22); const float* bd1 = IN(23); const float* Wd2 = IN(24); const float* bd2 = IN(25);
  const int TM_BASE[2] = {26, 42};   // tm_out_, tm_in_
  const int TA_BASE[2] = {58, 66};   // ta_out_, ta_in_
  const float* ptg = IN(74); const float* ptb = IN(75);
  const float* ptW1 = IN(76); const float* ptb1 = IN(77);
  const float* ptW2 = IN(78); const float* ptb2 = IN(79);

  float* ws = (float*)d_ws;
  const size_t PP = (size_t)LL * PDIM;
  float* z    = ws; ws += PP;
  float* t1   = ws; ws += PP;
  float* t2   = ws; ws += PP;
  float* t3   = ws; ws += PP;
  float* t4   = ws; ws += PP;
  float* pair = ws; ws += PP;
  float* biasProj = ws; ws += (size_t)LL * NHEADS;
  float* tbB  = ws; ws += (size_t)LL * HTRI;
  float* qB   = ws; ws += LSEQ * DMODEL;
  float* kB   = ws; ws += LSEQ * DMODEL;
  float* vB   = ws; ws += LSEQ * DMODEL;
  float* aO   = ws; ws += LSEQ * DMODEL;
  float* srcA = ws; ws += LSEQ * DMODEL;
  float* ffB  = ws; ws += LSEQ * 4 * DMODEL;
  float* sB   = ws; ws += LSEQ * 32;
  float* W2r  = ws; ws += 32 * 2048;
  float* Tbuf = ws; ws += 256 * 2048;

  float* srcOut  = (float*)d_out;            // (L, D)
  float* pairOut = srcOut + LSEQ * DMODEL;   // (L, L, P)

  // ---- pair bias for sequence attention ----
  lnorm(stream, pairwise, pw_g, pw_b, z, LL, PDIM);
  gemm(stream, z, PDIM, W_p2h, NHEADS, nullptr, nullptr, biasProj, NHEADS, LL, NHEADS, PDIM, 0);

  // ---- sequence MHA ----
  gemm(stream, src, DMODEL, Wq, DMODEL, bq, nullptr, qB, DMODEL, LSEQ, DMODEL, DMODEL, 0);
  gemm(stream, src, DMODEL, Wk, DMODEL, bk, nullptr, kB, DMODEL, LSEQ, DMODEL, DMODEL, 0);
  gemm(stream, src, DMODEL, Wv, DMODEL, bv, nullptr, vB, DMODEL, LSEQ, DMODEL, DMODEL, 0);
  mha_kernel<<<dim3(NHEADS, LSEQ / 16), 32, 0, stream>>>(qB, kB, vB, biasProj, mask, aO);
  gemm(stream, aO, DMODEL, Wo, DMODEL, bo, src, srcA, DMODEL, LSEQ, DMODEL, DMODEL, 0);
  lnorm(stream, srcA, n1g, n1b, qB, LSEQ, DMODEL);  // qB := src after LN1

  // ---- sequence transition ----
  gemm(stream, qB, DMODEL, stW1, 4 * DMODEL, stb1, nullptr, ffB, 4 * DMODEL, LSEQ, 4 * DMODEL, DMODEL, 1);
  gemm(stream, ffB, 4 * DMODEL, stW2, DMODEL, stb2, qB, srcA, DMODEL, LSEQ, DMODEL, 4 * DMODEL, 0);
  lnorm(stream, srcA, n2g, n2b, srcOut, LSEQ, DMODEL);  // final src output

  // ---- outer product mean (factored) ----
  gemm(stream, srcOut, DMODEL, Wd1, 32, bd1, nullptr, sB, 32, LSEQ, 32, DMODEL, 0);
  wd2_permute_kernel<<<(32 * 32 * 64) / 256, 256, 0, stream>>>(Wd2, W2r);
  gemm(stream, sB, 32, W2r, 2048, nullptr, nullptr, Tbuf, 2048, LSEQ, 2048, 32, 0);
  opm_finish_kernel<<<dim3(LSEQ, LSEQ / 16), 32, 0, stream>>>(sB, Tbuf, pairwise, bd2, pair);

  // ---- triangle multiplicative (outgoing, then ingoing) ----
  for (int m = 0; m < 2; m++) {
    int B = TM_BASE[m];
    lnorm(stream, pair, IN(B + 0), IN(B + 1), z, LL, PDIM);
    gemm(stream, z, PDIM, IN(B + 2), PDIM, IN(B + 8), nullptr, t1, PDIM, LL, PDIM, PDIM, 0);   // Wl
    gemm(stream, z, PDIM, IN(B + 4), PDIM, IN(B + 10), nullptr, t2, PDIM, LL, PDIM, PDIM, 0);  // Wlg
    gate(stream, t1, t1, t2, mask);
    gemm(stream, z, PDIM, IN(B + 3), PDIM, IN(B + 9), nullptr, t2, PDIM, LL, PDIM, PDIM, 0);   // Wr
    gemm(stream, z, PDIM, IN(B + 5), PDIM, IN(B + 11), nullptr, t3, PDIM, LL, PDIM, PDIM, 0);  // Wrg
    gate(stream, t2, t2, t3, mask);
    dim3 eg(LSEQ / 32, LSEQ / 32, PDIM);
    if (m == 0) tri_einsum_kernel<true><<<eg, 32, 0, stream>>>(t1, t2, t3);
    else        tri_einsum_kernel<false><<<eg, 32, 0, stream>>>(t1, t2, t3);
    lnorm(stream, t3, IN(B + 14), IN(B + 15), t4, LL, PDIM);
    gemm(stream, z, PDIM, IN(B + 6), PDIM, IN(B + 12), nullptr, t1, PDIM, LL, PDIM, PDIM, 0);  // Wog
    gate(stream, t4, t4, t1, nullptr);
    gemm(stream, t4, PDIM, IN(B + 7), PDIM, IN(B + 13), pair, pair, PDIM, LL, PDIM, PDIM, 0);  // Wo + residual
  }

  // ---- triangle attention (row, then column) ----
  for (int m = 0; m < 2; m++) {
    int B = TA_BASE[m];
    lnorm(stream, pair, IN(B + 0), IN(B + 1), z, LL, PDIM);
    const float* Wqkv = IN(B + 2);
    gemm(stream, z, PDIM, Wqkv + 0,        3 * PDIM, nullptr, nullptr, t1, PDIM, LL, PDIM, PDIM, 0);  // q
    gemm(stream, z, PDIM, Wqkv + PDIM,     3 * PDIM, nullptr, nullptr, t2, PDIM, LL, PDIM, PDIM, 0);  // k
    gemm(stream, z, PDIM, Wqkv + 2 * PDIM, 3 * PDIM, nullptr, nullptr, t3, PDIM, LL, PDIM, PDIM, 0);  // v
    gemm(stream, z, PDIM, IN(B + 3), HTRI, nullptr, nullptr, tbB, HTRI, LL, HTRI, PDIM, 0);           // bias
    dim3 ag(LSEQ, HTRI, LSEQ / 16);
    if (m == 0) tri_attn_kernel<true><<<ag, 32, 0, stream>>>(t1, t2, t3, tbB, mask, t4);
    else        tri_attn_kernel<false><<<ag, 32, 0, stream>>>(t1, t2, t3, tbB, mask, t4);
    gemm(stream, z, PDIM, IN(B + 4), PDIM, IN(B + 5), nullptr, t1, PDIM, LL, PDIM, PDIM, 0);          // Wg
    gate(stream, t4, t4, t1, nullptr);
    gemm(stream, t4, PDIM, IN(B + 6), PDIM, IN(B + 7), pair, pair, PDIM, LL, PDIM, PDIM, 0);          // Wo + residual
  }

  // ---- pair transition (fused) -> final pair output ----
  lnorm(stream, pair, ptg, ptb, z, LL, PDIM);
  pair_transition_kernel<<<LL / 16, 32, 0, stream>>>(z, ptW1, ptb1, ptW2, ptb2, pair, pairOut);
}
